// MoELoRALayer_1099511628253
// MI455X (gfx1250) — matrix operand order repaired and checked by reading, not compiled, not run on hardware
//
#include <hip/hip_runtime.h>
#include <hip/hip_bf16.h>

// ---------------------------------------------------------------------------
// MoE-LoRA fused implementation for gfx1250 (MI455X).
//   Stage 1: convert x, W_base to bf16; build Bcat (concat of B_stack ranks).
//   Stage 2: router + A-projection -> g[n,128] bf16 (combine * SCALING folded).
//   Stage 3: one bf16 WMMA GEMM, K = 4096 (+128 LoRA tail), f32 accum, bias
//            fused. Global->LDS staging via CDNA5 async-to-LDS (ASYNCcnt),
//            double-buffered LDS (explicit 2x unroll, compile-time buffer
//            index -> no MOVREL), one barrier per K-tile.
//   GEMM tiling: block 128x256, 8 waves (2x4), wave tile 64x64 ->
//                16 v_wmma per 16 ds_load_b128 per K-tile (matrix-unit bound).
// Workspace: Xbf 64MB + Wbf 32MB + Gbf 2MB + Bcat 1MB  (~99MB of d_ws).
// ---------------------------------------------------------------------------

typedef __bf16 bf16_t;
typedef __attribute__((ext_vector_type(8)))  __bf16 v8bf;
typedef __attribute__((ext_vector_type(16))) __bf16 v16bf;
typedef __attribute__((ext_vector_type(8)))  float  v8f;

#define D_IN  4096
#define D_OUT 4096
#define NTOK  8192
#define RANK  16
#define NEXP  8
#define KEXT  128          // NEXP*RANK
#define SCALING 2.0f       // 32/16

// --------------------------------------------------------- gfx1250 helpers
__device__ __forceinline__ unsigned lds_off32(const void* p) {
  // ISA 10.2: for LDS-aperture generic addresses, the LDS address is the low 32 bits of the pointer
  return (unsigned)(unsigned long long)p;
}

__device__ __forceinline__ void async_ld_b128(unsigned lds, const void* gp) {
  asm volatile("global_load_async_to_lds_b128 %0, %1, off"
               : : "v"(lds), "v"(gp) : "memory");
}

__device__ __forceinline__ void wait_async0() {
#if __has_builtin(__builtin_amdgcn_s_wait_asynccnt)
  __builtin_amdgcn_s_wait_asynccnt(0);
#else
  asm volatile("s_wait_asynccnt 0x0" ::: "memory");
#endif
}

// ---------------------------------------------------------------- converters
__global__ __launch_bounds__(256) void cvt_f32_bf16(const float4* __restrict__ src,
                                                    bf16_t* __restrict__ dst, int n4) {
  for (int i = blockIdx.x * 256 + threadIdx.x; i < n4; i += gridDim.x * 256) {
    float4 v = src[i];
    bf16_t o[4] = {(bf16_t)v.x, (bf16_t)v.y, (bf16_t)v.z, (bf16_t)v.w};
    *reinterpret_cast<uint2*>(dst + (size_t)i * 4) = *reinterpret_cast<uint2*>(o);
  }
}

// Bcat[o][e*16+r] = B_stack[e][o][r]   (4096 x 128, row-major)
__global__ __launch_bounds__(256) void build_bcat(const float* __restrict__ B,
                                                  bf16_t* __restrict__ Bc) {
  int i = blockIdx.x * 256 + threadIdx.x;           // 0 .. 4096*128-1
  if (i < D_OUT * KEXT) {
    int o = i >> 7, j = i & 127;
    int e = j >> 4, r = j & 15;
    Bc[i] = (bf16_t)B[((size_t)e * D_OUT + o) * RANK + r];
  }
}

// ------------------------------------------------------------------- router
__global__ __launch_bounds__(256) void router_kernel(const float* __restrict__ x,
                                                     const float* __restrict__ Wr,
                                                     const float* __restrict__ A,
                                                     bf16_t* __restrict__ g) {
  __shared__ float sx[D_IN];
  __shared__ float hv[8 + KEXT];
  __shared__ float cw[NEXP];
  const int n = blockIdx.x;
  const int tid = threadIdx.x;
  const float* xr = x + (size_t)n * D_IN;
  for (int i = tid; i < D_IN; i += 256) sx[i] = xr[i];
  __syncthreads();

  const int wid = tid >> 5, lane = tid & 31;
  for (int j = wid; j < 8 + KEXT; j += 8) {
    const float* row = (j < 8) ? (Wr + (size_t)j * D_IN)
                               : (A + (size_t)(j - 8) * D_IN);
    float s = 0.f;
    for (int k = lane; k < D_IN; k += 32) s += sx[k] * row[k];
    #pragma unroll
    for (int off = 16; off; off >>= 1) s += __shfl_xor(s, off, 32);
    if (lane == 0) hv[j] = s;
  }
  __syncthreads();

  if (tid == 0) {
    int i1 = 0; float l1 = hv[0];
    #pragma unroll
    for (int e = 1; e < NEXP; ++e) if (hv[e] > l1) { l1 = hv[e]; i1 = e; }
    int i2 = (i1 == 0) ? 1 : 0; float l2 = hv[i2];
    #pragma unroll
    for (int e = 0; e < NEXP; ++e)
      if (e != i1 && hv[e] > l2) { l2 = hv[e]; i2 = e; }
    float w1 = 1.f / (1.f + __expf(l2 - l1));   // renormalized top-2 softmax
    #pragma unroll
    for (int e = 0; e < NEXP; ++e) cw[e] = 0.f;
    cw[i1] = w1; cw[i2] = 1.f - w1;
  }
  __syncthreads();

  if (tid < KEXT) {
    float v = cw[tid >> 4] * SCALING * hv[8 + tid];
    g[(size_t)n * KEXT + tid] = (bf16_t)v;
  }
}

// --------------------------------------------------------------------- GEMM
// out[m][o] = bias[o] + sum_k Xbf[m][k]*Wbf[o][k] + sum_j Gbf[m][j]*Bcat[o][j]
// Block tile 128x256, BK=32, 8 waves in 2x4 grid, wave tile 64x64 (4x4 frags).
#define BM 128
#define BN 256
#define BK 32
#define LDSS 40            // LDS row stride (bf16 elems): 32 + 8 pad, 80B

__global__ __launch_bounds__(256)
void moe_gemm_kernel(const bf16_t* __restrict__ Xbf, const bf16_t* __restrict__ Wbf,
                     const bf16_t* __restrict__ Gbf, const bf16_t* __restrict__ Bcat,
                     const float* __restrict__ bias, float* __restrict__ out) {
  __shared__ bf16_t As[2][BM * LDSS];   // 2 x 10 KB
  __shared__ bf16_t Bs[2][BN * LDSS];   // 2 x 20 KB

  const int tid  = threadIdx.x;
  const int m0   = blockIdx.x * BM;
  const int n0   = blockIdx.y * BN;
  const int wid  = tid >> 5;
  const int lane = tid & 31;
  const int wm   = wid >> 2;        // 0..1 -> 64 rows each
  const int wn   = wid & 3;         // 0..3 -> 64 cols each
  const int r0   = tid >> 2;        // staging row 0..63
  const int c4   = tid & 3;         // staging 16B chunk within 64B row

  v8f acc[4][4];
  #pragma unroll
  for (int a = 0; a < 4; ++a)
    #pragma unroll
    for (int b = 0; b < 4; ++b) acc[a][b] = (v8f)0.f;

  const int NT_MAIN = D_IN / BK;            // 128
  const int NT_ALL  = NT_MAIN + KEXT / BK;  // 132 (even)

  // Per-thread LDS destinations, one flat array per buffer (constant-indexed).
  unsigned ldsA0[2], ldsA1[2], ldsB0[4], ldsB1[4];
  #pragma unroll
  for (int p = 0; p < 2; ++p) {
    ldsA0[p] = lds_off32(&As[0][(r0 + p * 64) * LDSS + c4 * 8]);
    ldsA1[p] = lds_off32(&As[1][(r0 + p * 64) * LDSS + c4 * 8]);
  }
  #pragma unroll
  for (int p = 0; p < 4; ++p) {
    ldsB0[p] = lds_off32(&Bs[0][(r0 + p * 64) * LDSS + c4 * 8]);
    ldsB1[p] = lds_off32(&Bs[1][(r0 + p * 64) * LDSS + c4 * 8]);
  }

  auto issue = [&](int kt, const unsigned (&lA)[2], const unsigned (&lB)[4]) {
    const bf16_t *ap, *bp; int lda, ldb, kk;
    if (kt < NT_MAIN) { ap = Xbf + (size_t)m0 * D_IN; lda = D_IN;
                        bp = Wbf + (size_t)n0 * D_IN; ldb = D_IN;
                        kk = kt * BK; }
    else              { ap = Gbf + (size_t)m0 * KEXT; lda = KEXT;
                        bp = Bcat + (size_t)n0 * KEXT; ldb = KEXT;
                        kk = (kt - NT_MAIN) * BK; }
    #pragma unroll
    for (int p = 0; p < 2; ++p)
      async_ld_b128(lA[p], ap + (size_t)(r0 + p * 64) * lda + kk + c4 * 8);
    #pragma unroll
    for (int p = 0; p < 4; ++p)
      async_ld_b128(lB[p], bp + (size_t)(r0 + p * 64) * ldb + kk + c4 * 8);
  };

  // ISA 7.12.2 bf16 fragment lane layouts:
  //  A 16x32: lane<16 holds K[0..7],K[16..23]; lane>=16 holds K[8..15],K[24..31]
  //  B 32x16: lane<16 holds K[0..7],K[16..23]; lane>=16 holds K[8..15],K[24..31]
  const int aoff = (lane >> 4) * 8;   // bf16 elems
  const int boff = (lane >> 4) * 8;
  const int arow = wm * 64;
  const int brow = wn * 64;
  const int lrow = lane & 15;

  auto compute = [&](const bf16_t* __restrict__ Ab, const bf16_t* __restrict__ Bb) {
    v16bf bfrag[4];
    #pragma unroll
    for (int fn = 0; fn < 4; ++fn) {
      int row = brow + fn * 16 + lrow;
      v8bf lo = *reinterpret_cast<const v8bf*>(&Bb[row * LDSS + boff]);
      v8bf hi = *reinterpret_cast<const v8bf*>(&Bb[row * LDSS + boff + 16]);
      bfrag[fn] = __builtin_shufflevector(lo, hi, 0,1,2,3,4,5,6,7,8,9,10,11,12,13,14,15);
    }
    #pragma unroll
    for (int fm = 0; fm < 4; ++fm) {
      int row = arow + fm * 16 + lrow;
      v8bf lo = *reinterpret_cast<const v8bf*>(&Ab[row * LDSS + aoff]);
      v8bf hi = *reinterpret_cast<const v8bf*>(&Ab[row * LDSS + 16 + aoff]);
      v16bf afrag = __builtin_shufflevector(lo, hi, 0,1,2,3,4,5,6,7,8,9,10,11,12,13,14,15);
      #pragma unroll
      for (int fn = 0; fn < 4; ++fn)
        acc[fm][fn] = __builtin_amdgcn_wmma_f32_16x16x32_bf16(
            false, afrag, false, bfrag[fn],
            (short)0, acc[fm][fn], false, false);
    }
  };

  issue(0, ldsA0, ldsB0);
  wait_async0();
  __syncthreads();

  // Explicit 2x unroll so each half uses a compile-time LDS buffer.
  for (int kt = 0; kt < NT_ALL; kt += 2) {
    issue(kt + 1, ldsA1, ldsB1);            // next tile -> buffer 1
    compute(As[0], Bs[0]);
    wait_async0();
    __syncthreads();

    if (kt + 2 < NT_ALL) issue(kt + 2, ldsA0, ldsB0);  // -> buffer 0
    compute(As[1], Bs[1]);
    if (kt + 2 < NT_ALL) {
      wait_async0();
      __syncthreads();
    }
  }

  // Epilogue: C/D layout — lane<16: N=lane, M=r; lane>=16: N=lane-16, M=8+r
  const int col  = lane & 15;
  const int rsel = (lane >> 4) * 8;
  #pragma unroll
  for (int fm = 0; fm < 4; ++fm) {
    int mbase = m0 + wm * 64 + fm * 16 + rsel;
    #pragma unroll
    for (int fn = 0; fn < 4; ++fn) {
      int nidx = n0 + wn * 64 + fn * 16 + col;
      float bz = bias[nidx];
      #pragma unroll
      for (int r = 0; r < 8; ++r)
        out[(size_t)(mbase + r) * D_OUT + nidx] = acc[fm][fn][r] + bz;
    }
  }
}

// ------------------------------------------------------------------- launch
extern "C" void kernel_launch(void* const* d_in, const int* in_sizes, int n_in,
                              void* d_out, int out_size, void* d_ws, size_t ws_size,
                              hipStream_t stream) {
  const float* x   = (const float*)d_in[0];  // [4,2048,4096]
  const float* Wb  = (const float*)d_in[1];  // [4096,4096]
  const float* bb  = (const float*)d_in[2];  // [4096]
  const float* Wr  = (const float*)d_in[3];  // [8,4096]
  const float* Ast = (const float*)d_in[4];  // [8,16,4096]
  const float* Bst = (const float*)d_in[5];  // [8,4096,16]
  float* out = (float*)d_out;

  char* ws = (char*)d_ws;
  bf16_t* Xbf = (bf16_t*)ws;                         // 8192*4096
  bf16_t* Wbf = Xbf + (size_t)NTOK * D_IN;           // 4096*4096
  bf16_t* Gbf = Wbf + (size_t)D_OUT * D_IN;          // 8192*128
  bf16_t* Bc  = Gbf + (size_t)NTOK * KEXT;           // 4096*128

  cvt_f32_bf16<<<4096, 256, 0, stream>>>((const float4*)x,  Xbf, NTOK * D_IN / 4);
  cvt_f32_bf16<<<4096, 256, 0, stream>>>((const float4*)Wb, Wbf, D_OUT * D_IN / 4);
  build_bcat  <<<(D_OUT * KEXT + 255) / 256, 256, 0, stream>>>(Bst, Bc);
  router_kernel<<<NTOK, 256, 0, stream>>>(x, Wr, Ast, Gbf);

  dim3 grid(NTOK / BM, D_OUT / BN);   // 64 x 16
  moe_gemm_kernel<<<grid, 256, 0, stream>>>(Xbf, Wbf, Gbf, Bc, bb, out);
}
